// Attention_LR_65249143160949
// MI455X (gfx1250) — matrix operand order repaired and checked by reading, not compiled, not run on hardware
//
#include <hip/hip_runtime.h>
#include <hip/hip_bf16.h>

typedef __attribute__((ext_vector_type(16))) _Float16 v16h;
typedef __attribute__((ext_vector_type(8)))  _Float16 v8h;
typedef __attribute__((ext_vector_type(8)))  float    v8f;

#define B_    4
#define C_    512
#define NPIX  2304
#define HEADS 8
#define DH    64
#define INNER 512
#define CTXD  768
#define CTXL  77
#define JTOT  (CTXL + 1 + NPIX)   /* 2382 valid keys */
#define JP    2432                /* padded to 38*64 */
#define NJT   (JP / 64)           /* 38 key tiles */
#define LN_EPS 1e-5f
#define QSCALE 0.125f             /* dh^-0.5 = 1/8 */

__device__ __forceinline__ v8f zero8() {
  v8f z = {0.f,0.f,0.f,0.f,0.f,0.f,0.f,0.f};
  return z;
}

__device__ __forceinline__ v8f wmma16(v16h a, v16h b, v8f c) {
  return __builtin_amdgcn_wmma_f32_16x16x32_f16(false, a, false, b, (short)0, c, false, false);
}

// ---- CDNA5 async global->LDS copy (no VGPR transit, tracked by ASYNCcnt) ---
// GVS form: mem_addr = SGPR_base + VGPR_offset ; LDS dest = low 32 bits of the
// generic shared pointer (flat->LDS truncates to addr[31:0]).
__device__ __forceinline__ void async_copy_b128(const void* lds_dst,
                                                const void* gbase,
                                                unsigned byte_off) {
  unsigned lds_addr = (unsigned)(size_t)lds_dst;
  asm volatile("global_load_async_to_lds_b128 %0, %1, %2"
               :
               : "v"(lds_addr), "v"(byte_off), "s"((unsigned long long)gbase)
               : "memory");
}
__device__ __forceinline__ void async_wait0() {
  asm volatile("s_wait_asynccnt 0" ::: "memory");
}

__device__ __forceinline__ void blockReduce2_128(float& a, float& b, float* red) {
#pragma unroll
  for (int m = 1; m < 32; m <<= 1) { a += __shfl_xor(a, m); b += __shfl_xor(b, m); }
  int w = threadIdx.x >> 5;
  if ((threadIdx.x & 31) == 0) { red[w] = a; red[4 + w] = b; }
  __syncthreads();
  a = red[0] + red[1] + red[2] + red[3];
  b = red[4] + red[5] + red[6] + red[7];
  __syncthreads();
}

// ---------------- weight conversion to f16 (fused QKV weight + Wout) -------
__global__ void cvtw_kernel(const float* __restrict__ Wq,
                            const float* __restrict__ Wkv,
                            const float* __restrict__ Wout,
                            _Float16* __restrict__ Wf,
                            _Float16* __restrict__ Wo) {
  int idx = blockIdx.x * 256 + threadIdx.x;
  const int t1 = C_ * 640;
  if (idx < t1) {
    int k = idx / 640, n = idx - k * 640;
    float val = (n < 512) ? Wq[(size_t)k * 512 + n] : Wkv[(size_t)k * 128 + (n - 512)];
    Wf[idx] = (_Float16)val;
  } else if (idx < t1 + 512 * 512) {
    int i2 = idx - t1;
    Wo[i2] = (_Float16)Wout[i2];
  }
}

// ---------------- LayerNorm of x (transposed read) -> xn f16 ----------------
__global__ __launch_bounds__(128) void ln_x_kernel(const float* __restrict__ x,
                                                   const float* __restrict__ g,
                                                   _Float16* __restrict__ xn) {
  __shared__ float red[8];
  const int row = blockIdx.x;          // b*NPIX + n
  const int b = row / NPIX;
  const int n = row - b * NPIX;
  const float* xb = x + (size_t)b * C_ * NPIX;
  float v[4], s = 0.f, s2 = 0.f;
#pragma unroll
  for (int i = 0; i < 4; ++i) {
    int c = threadIdx.x + i * 128;
    v[i] = xb[(size_t)c * NPIX + n];
    s += v[i]; s2 += v[i] * v[i];
  }
  blockReduce2_128(s, s2, red);
  float mu = s * (1.0f / 512.0f);
  float var = s2 * (1.0f / 512.0f) - mu * mu;
  float rs = rsqrtf(var + LN_EPS);
#pragma unroll
  for (int i = 0; i < 4; ++i) {
    int c = threadIdx.x + i * 128;
    xn[(size_t)row * C_ + c] = (_Float16)((v[i] - mu) * rs * g[c]);
  }
}

// ---------------- context LN + projection -> ck/cv (tiny) -------------------
__global__ __launch_bounds__(128) void ctx_kernel(const float* __restrict__ ctx,
                                                  const float* __restrict__ g,
                                                  const float* __restrict__ be,
                                                  const float* __restrict__ Wctx,
                                                  const float* __restrict__ bctx,
                                                  _Float16* __restrict__ kf,
                                                  _Float16* __restrict__ vf) {
  __shared__ float cn[CTXD];
  __shared__ float red[8];
  const int bj = blockIdx.x;           // b*CTXL + j
  const int b = bj / CTXL;
  const int j = bj - b * CTXL;
  const float* src = ctx + (size_t)bj * CTXD;
  float v[6], s = 0.f, s2 = 0.f;
#pragma unroll
  for (int i = 0; i < 6; ++i) {
    int c = threadIdx.x + i * 128;
    v[i] = src[c]; s += v[i]; s2 += v[i] * v[i];
  }
  blockReduce2_128(s, s2, red);
  float mu = s * (1.0f / 768.0f);
  float var = s2 * (1.0f / 768.0f) - mu * mu;
  float rs = rsqrtf(var + LN_EPS);
#pragma unroll
  for (int i = 0; i < 6; ++i) {
    int c = threadIdx.x + i * 128;
    cn[c] = (v[i] - mu) * rs * g[c] + be[c];
  }
  __syncthreads();
  float acc = bctx[threadIdx.x];
  for (int k = 0; k < CTXD; ++k) acc += cn[k] * Wctx[(size_t)k * 128 + threadIdx.x];
  if (threadIdx.x < 64) kf[((size_t)b * JP + j) * DH + threadIdx.x] = (_Float16)acc;
  else                  vf[((size_t)b * JP + j) * DH + (threadIdx.x - 64)] = (_Float16)acc;
}

// ---------------- null KV at j=77 and zero-fill pad keys --------------------
__global__ void nullpad_kernel(const float* __restrict__ nkv,
                               _Float16* __restrict__ kf,
                               _Float16* __restrict__ vf) {
  int idx = blockIdx.x * 256 + threadIdx.x;
  const int total = B_ * 51 * 64;      // slot0 = null kv, slots 1..50 = pad rows
  if (idx >= total) return;
  int d = idx & 63;
  int t = idx >> 6;
  int slot = t % 51;
  int b = t / 51;
  int j = (slot == 0) ? CTXL : (JTOT - 1 + slot);   // 77, or 2382..2431
  _Float16 kvv = (slot == 0) ? (_Float16)nkv[d]      : (_Float16)0.f;
  _Float16 vvv = (slot == 0) ? (_Float16)nkv[64 + d] : (_Float16)0.f;
  kf[((size_t)b * JP + j) * DH + d] = kvv;
  vf[((size_t)b * JP + j) * DH + d] = vvv;
}

// ---------------- WMMA GEMM: EPI=0 -> fused QKV routing, EPI=1 -> f32 out ---
// Double-buffered LDS tiles: async A copies + register-held B tile for the
// next k-step are issued before computing on the current buffers.
template<int EPI>
__global__ __launch_bounds__(256) void gemm_kernel(const _Float16* __restrict__ A,
                                                   const _Float16* __restrict__ Bw,
                                                   int ldB,
                                                   _Float16* __restrict__ qout,
                                                   _Float16* __restrict__ kout,
                                                   _Float16* __restrict__ vout,
                                                   float* __restrict__ fout) {
  __shared__ __align__(16) _Float16 As[2][64][40];   // row-major [m][k]
  __shared__ __align__(16) _Float16 Bs[2][64][40];   // transposed [n][k]
  const int tid = threadIdx.x;
  const int lane = tid & 31;
  const int w = tid >> 5;
  const int wm = w & 3, wn = w >> 2;
  const int lo16 = lane & 15;
  const int isHi = (lane >= 16);
  const int nb = blockIdx.x * 64;
  const int mb = blockIdx.y * 64;

  v8f acc[2] = {zero8(), zero8()};
  const int ar = tid >> 2, ac = (tid & 3) * 8;    // A staging: 64x32 halves
  const int bk = tid >> 3, bn = (tid & 7) * 8;    // B staging: 32x64 halves

  auto stageA = [&](int kb, int buf) {
    async_copy_b128(&As[buf][ar][ac],
                    A + (size_t)mb * 512 + kb,
                    (unsigned)((ar * 512 + ac) * 2));
  };
  auto loadB = [&](int kb) -> v8h {
    return *(const v8h*)(Bw + ((size_t)(kb + bk)) * ldB + nb + bn);
  };
  auto storeB = [&](int buf, v8h bv) {
#pragma unroll
    for (int e = 0; e < 8; ++e) Bs[buf][bn + e][bk] = bv[e];
  };

  // prologue: stage first tile pair
  stageA(0, 0);
  storeB(0, loadB(0));
  async_wait0();
  __syncthreads();

  for (int it = 0; it < 16; ++it) {
    const int kb = it * 32;
    const int cur = it & 1, nxt = cur ^ 1;
    const bool more = (it + 1 < 16);
    v8h bnx = {};
    if (more) {                       // issue next tile while computing current
      stageA(kb + 32, nxt);
      bnx = loadB(kb + 32);
      if (kb + 64 < 512)
        __builtin_prefetch(Bw + ((size_t)(kb + 64 + bk)) * ldB + nb + bn, 0, 1);
    }

    union { v16h v; v8h h[2]; } a;
    const _Float16* ap = &As[cur][wm * 16 + lo16][isHi ? 8 : 0];
    a.h[0] = *(const v8h*)ap;
    a.h[1] = *(const v8h*)(ap + 16);
#pragma unroll
    for (int t = 0; t < 2; ++t) {
      union { v16h v; v8h h[2]; } bb;
      const _Float16* bp = &Bs[cur][wn * 32 + t * 16 + lo16][isHi ? 8 : 0];
      bb.h[0] = *(const v8h*)bp;
      bb.h[1] = *(const v8h*)(bp + 16);
      acc[t] = wmma16(a.v, bb.v, acc[t]);
    }

    if (more) {
      storeB(nxt, bnx);
      async_wait0();
    }
    __syncthreads();
  }

#pragma unroll
  for (int t = 0; t < 2; ++t) {
    const int gn = nb + wn * 32 + t * 16 + lo16;
#pragma unroll
    for (int r = 0; r < 8; ++r) {
      const int gm = mb + wm * 16 + (isHi ? 8 + r : r);
      float d = acc[t][r];
      if (EPI == 0) {
        int bi = gm / NPIX;
        int np = gm - bi * NPIX;
        if (gn < 512) {
          qout[(size_t)gm * INNER + gn] = (_Float16)(d * QSCALE);
        } else if (gn < 576) {
          kout[((size_t)bi * JP + (CTXL + 1) + np) * DH + (gn - 512)] = (_Float16)d;
        } else {
          vout[((size_t)bi * JP + (CTXL + 1) + np) * DH + (gn - 576)] = (_Float16)d;
        }
      } else {
        fout[(size_t)gm * 512 + gn] = d;
      }
    }
  }
}

// ---------------- flash attention: 128 q-rows/block, 16 rows/wave -----------
// Double-buffered K/V tiles: next K async-copies + next V global loads are
// issued before computing on the current tile; V is transposed into LDS after
// the current tile's WMMAs complete.
__global__ __launch_bounds__(256) void attn_kernel(const _Float16* __restrict__ qf,
                                                   const _Float16* __restrict__ kf,
                                                   const _Float16* __restrict__ vf,
                                                   _Float16* __restrict__ of) {
  __shared__ __align__(16) _Float16 Qs[128][72];         // [m][d]
  __shared__ __align__(16) _Float16 Ks[2][64][72];       // [j][d]
  __shared__ __align__(16) _Float16 Vt[2][64][72];       // transposed [d][j]
  __shared__ __align__(16) _Float16 Ps[8][16][72];       // per-wave P tile [m][j]

  const int tid  = threadIdx.x;
  const int lane = tid & 31;
  const int w    = tid >> 5;
  const int lo16 = lane & 15;
  const int isHi = (lane >= 16);
  const int b    = blockIdx.x >> 3;
  const int h    = blockIdx.x & 7;
  const int m0   = blockIdx.y * 128;

  auto stageK = [&](int jt, int buf) {
    const _Float16* kbase = kf + ((size_t)(b * JP + jt * 64)) * DH;
#pragma unroll
    for (int i = 0; i < 2; ++i) {
      int ch = tid + 256 * i;
      int r = ch >> 3, cg = (ch & 7) * 8;
      async_copy_b128(&Ks[buf][r][cg], kbase, (unsigned)((r * DH + cg) * 2));
    }
  };
  auto loadV = [&](int jt, v8h* vr) {
#pragma unroll
    for (int i = 0; i < 2; ++i) {
      int ch = tid + 256 * i;
      int j = ch >> 3, dg = (ch & 7) * 8;
      vr[i] = *(const v8h*)(vf + ((size_t)(b * JP + jt * 64 + j)) * DH + dg);
    }
  };
  auto storeV = [&](int buf, const v8h* vr) {
#pragma unroll
    for (int i = 0; i < 2; ++i) {
      int ch = tid + 256 * i;
      int j = ch >> 3, dg = (ch & 7) * 8;
#pragma unroll
      for (int e = 0; e < 8; ++e) Vt[buf][dg + e][j] = vr[i][e];
    }
  };

  // prologue: stage Q tile (async) + first K/V tiles
  {
    const _Float16* qbase = qf + ((size_t)(b * NPIX + m0)) * INNER + h * DH;
#pragma unroll
    for (int i = 0; i < 4; ++i) {
      int ch = tid + 256 * i;
      int r = ch >> 3, cg = (ch & 7) * 8;
      async_copy_b128(&Qs[r][cg], qbase, (unsigned)((r * INNER + cg) * 2));
    }
  }
  stageK(0, 0);
  {
    v8h v0[2];
    loadV(0, v0);
    storeV(0, v0);
  }
  async_wait0();
  __syncthreads();

  v8f oa[4] = {zero8(), zero8(), zero8(), zero8()};
  float mrow[8], lrow[8];
#pragma unroll
  for (int r = 0; r < 8; ++r) { mrow[r] = -1e30f; lrow[r] = 0.f; }

  for (int jt = 0; jt < NJT; ++jt) {
    const int jb = jt * 64;
    const int cur = jt & 1, nxt = cur ^ 1;
    const bool more = (jt + 1 < NJT);
    v8h vnx[2];
    if (more) {                      // issue next tile while computing current
      stageK(jt + 1, nxt);
      loadV(jt + 1, vnx);
    }

    // sim tile: 16 rows x 64 keys per wave, dh=64 in two K=32 steps
    v8f s[4] = {zero8(), zero8(), zero8(), zero8()};
#pragma unroll
    for (int ks = 0; ks < 2; ++ks) {
      union { v16h v; v8h h[2]; } a;
      const _Float16* ap = &Qs[w * 16 + lo16][ks * 32 + (isHi ? 8 : 0)];
      a.h[0] = *(const v8h*)ap;
      a.h[1] = *(const v8h*)(ap + 16);
#pragma unroll
      for (int nt = 0; nt < 4; ++nt) {
        union { v16h v; v8h h[2]; } bb;
        const _Float16* bp = &Ks[cur][nt * 16 + lo16][ks * 32 + (isHi ? 8 : 0)];
        bb.h[0] = *(const v8h*)bp;
        bb.h[1] = *(const v8h*)(bp + 16);
        s[nt] = wmma16(a.v, bb.v, s[nt]);
      }
    }

    // mask out-of-range keys
#pragma unroll
    for (int nt = 0; nt < 4; ++nt) {
      if (jb + nt * 16 + lo16 >= JTOT) {
#pragma unroll
        for (int r = 0; r < 8; ++r) s[nt][r] = -1e30f;
      }
    }

    // online softmax: per-row reductions are 16-lane (half-wave) shuffles
#pragma unroll
    for (int r = 0; r < 8; ++r) {
      float v = fmaxf(fmaxf(s[0][r], s[1][r]), fmaxf(s[2][r], s[3][r]));
      v = fmaxf(v, __shfl_xor(v, 1));
      v = fmaxf(v, __shfl_xor(v, 2));
      v = fmaxf(v, __shfl_xor(v, 4));
      v = fmaxf(v, __shfl_xor(v, 8));
      float mn = fmaxf(mrow[r], v);
      float alpha = __expf(mrow[r] - mn);
      mrow[r] = mn;
      lrow[r] *= alpha;
      oa[0][r] *= alpha; oa[1][r] *= alpha; oa[2][r] *= alpha; oa[3][r] *= alpha;
      float sum = 0.f;
#pragma unroll
      for (int nt = 0; nt < 4; ++nt) {
        float p = __expf(s[nt][r] - mn);
        s[nt][r] = p;
        sum += p;
      }
      sum += __shfl_xor(sum, 1);
      sum += __shfl_xor(sum, 2);
      sum += __shfl_xor(sum, 4);
      sum += __shfl_xor(sum, 8);
      lrow[r] += sum;
    }

    // write P tile (f16) to this wave's LDS staging, re-load in A layout
    {
      const int rbase = isHi ? 8 : 0;
#pragma unroll
      for (int r = 0; r < 8; ++r)
#pragma unroll
        for (int nt = 0; nt < 4; ++nt)
          Ps[w][rbase + r][nt * 16 + lo16] = (_Float16)s[nt][r];
    }

    // O += P @ V
#pragma unroll
    for (int ks = 0; ks < 2; ++ks) {
      union { v16h v; v8h h[2]; } a;
      const _Float16* ap = &Ps[w][lo16][ks * 32 + (isHi ? 8 : 0)];
      a.h[0] = *(const v8h*)ap;
      a.h[1] = *(const v8h*)(ap + 16);
#pragma unroll
      for (int nt = 0; nt < 4; ++nt) {
        union { v16h v; v8h h[2]; } bb;
        const _Float16* bp = &Vt[cur][nt * 16 + lo16][ks * 32 + (isHi ? 8 : 0)];
        bb.h[0] = *(const v8h*)bp;
        bb.h[1] = *(const v8h*)(bp + 16);
        oa[nt] = wmma16(a.v, bb.v, oa[nt]);
      }
    }

    if (more) {
      storeV(nxt, vnx);
      async_wait0();
    }
    __syncthreads();
  }

#pragma unroll
  for (int r = 0; r < 8; ++r) {
    float inv = 1.0f / lrow[r];
    int gm = m0 + w * 16 + (isHi ? 8 + r : r);
    size_t base = ((size_t)(b * NPIX + gm)) * INNER + h * DH;
    of[base +  0 + lo16] = (_Float16)(oa[0][r] * inv);
    of[base + 16 + lo16] = (_Float16)(oa[1][r] * inv);
    of[base + 32 + lo16] = (_Float16)(oa[2][r] * inv);
    of[base + 48 + lo16] = (_Float16)(oa[3][r] * inv);
  }
}

// ---------------- final LN + residual + transpose back ----------------------
__global__ __launch_bounds__(128) void final_kernel(const float* __restrict__ tmp,
                                                    const float* __restrict__ x,
                                                    const float* __restrict__ g,
                                                    float* __restrict__ out) {
  __shared__ float red[8];
  const int row = blockIdx.x;
  const int b = row / NPIX;
  const int n = row - b * NPIX;
  float v[4], s = 0.f, s2 = 0.f;
#pragma unroll
  for (int i = 0; i < 4; ++i) {
    int c = threadIdx.x + i * 128;
    v[i] = tmp[(size_t)row * C_ + c];
    s += v[i]; s2 += v[i] * v[i];
  }
  blockReduce2_128(s, s2, red);
  float mu = s * (1.0f / 512.0f);
  float var = s2 * (1.0f / 512.0f) - mu * mu;
  float rs = rsqrtf(var + LN_EPS);
#pragma unroll
  for (int i = 0; i < 4; ++i) {
    int c = threadIdx.x + i * 128;
    size_t gi = ((size_t)b * C_ + c) * NPIX + n;
    out[gi] = x[gi] + (v[i] - mu) * rs * g[c];
  }
}

extern "C" void kernel_launch(void* const* d_in, const int* in_sizes, int n_in,
                              void* d_out, int out_size, void* d_ws, size_t ws_size,
                              hipStream_t stream) {
  (void)in_sizes; (void)n_in; (void)out_size; (void)ws_size;
  const float* x       = (const float*)d_in[0];
  const float* context = (const float*)d_in[1];
  const float* norm_g  = (const float*)d_in[2];
  const float* null_kv = (const float*)d_in[3];
  const float* Wq      = (const float*)d_in[4];
  const float* Wkv     = (const float*)d_in[5];
  const float* ctx_g   = (const float*)d_in[6];
  const float* ctx_b   = (const float*)d_in[7];
  const float* Wctx    = (const float*)d_in[8];
  const float* bctx    = (const float*)d_in[9];
  const float* Wout    = (const float*)d_in[10];
  const float* out_g   = (const float*)d_in[11];
  float* out = (float*)d_out;

  char* ws = (char*)d_ws;
  size_t off = 0;
  auto take = [&](size_t bytes) -> char* {
    char* p = ws + off;
    off += (bytes + 255) & ~(size_t)255;
    return p;
  };
  _Float16* xn  = (_Float16*)take((size_t)B_ * NPIX * C_ * 2);      // 9.4 MB
  _Float16* qf  = (_Float16*)take((size_t)B_ * NPIX * INNER * 2);   // 9.4 MB
  _Float16* kf  = (_Float16*)take((size_t)B_ * JP * DH * 2);        // 1.2 MB
  _Float16* vf  = (_Float16*)take((size_t)B_ * JP * DH * 2);        // 1.2 MB
  _Float16* of  = (_Float16*)take((size_t)B_ * NPIX * INNER * 2);   // 9.4 MB
  _Float16* Wf  = (_Float16*)take((size_t)C_ * 640 * 2);            // 0.7 MB
  _Float16* Wo  = (_Float16*)take((size_t)INNER * C_ * 2);          // 0.5 MB
  float*    tmp = (float*)take((size_t)B_ * NPIX * C_ * 4);         // 18.9 MB

  cvtw_kernel<<<dim3((C_ * 640 + 512 * 512 + 255) / 256), 256, 0, stream>>>(Wq, Wkv, Wout, Wf, Wo);
  ln_x_kernel<<<dim3(B_ * NPIX), 128, 0, stream>>>(x, norm_g, xn);
  ctx_kernel<<<dim3(B_ * CTXL), 128, 0, stream>>>(context, ctx_g, ctx_b, Wctx, bctx, kf, vf);
  nullpad_kernel<<<dim3((B_ * 51 * 64 + 255) / 256), 256, 0, stream>>>(null_kv, kf, vf);
  gemm_kernel<0><<<dim3(10, (B_ * NPIX) / 64), 256, 0, stream>>>(xn, Wf, 640, qf, kf, vf, nullptr);
  attn_kernel<<<dim3(B_ * HEADS, NPIX / 128), 256, 0, stream>>>(qf, kf, vf, of);
  gemm_kernel<1><<<dim3(8, (B_ * NPIX) / 64), 256, 0, stream>>>(of, Wo, 512, nullptr, nullptr, nullptr, tmp);
  final_kernel<<<dim3(B_ * NPIX), 128, 0, stream>>>(tmp, x, out_g, out);
}
